// ViTMultiHeadAttention_13950053778156
// MI455X (gfx1250) — compile-verified, matrix operand order checked by reading, not run on hardware
//
#include <hip/hip_runtime.h>
#include <hip/hip_bf16.h>

typedef __attribute__((ext_vector_type(16))) _Float16     v16h;
typedef __attribute__((ext_vector_type(8)))  float        v8f;
typedef __attribute__((ext_vector_type(4)))  unsigned int u32x4;
typedef __attribute__((ext_vector_type(8)))  unsigned int u32x8;

// ---------------------------------------------------------------------------
// WMMA fragment gather from an LDS tile.
// 16-bit A-matrix 16x32 lane layout (CDNA5 ISA 7.12.2):
//   lanes 0-15 : row M = lane,    VGPR v holds K pair {k0 + (v>>2)*16 + (v&3)*2}
//   lanes 16-31: row M = lane-16, kb = +8
// B fragments come from tiles stored transposed ([N][K]) so the same dword
// gather applies (lane -> N, VGPR -> K pair); compiler fuses to ds_load_b128.
// ---------------------------------------------------------------------------
__device__ __forceinline__ v16h load_frag(const _Float16* __restrict__ base,
                                          int row_off, int stride, int k0, int lane) {
  union { v16h h; unsigned int u[8]; } f;
  const int m  = row_off + (lane & 15);
  const int kb = k0 + ((lane >> 4) << 3);
  const _Float16* p = base + m * stride;
#pragma unroll
  for (int v = 0; v < 8; ++v) {
    const int k = kb + ((v >> 2) << 4) + ((v & 3) << 1);
    f.u[v] = *(const unsigned int*)(p + k);
  }
  return f.h;
}

__device__ __forceinline__ v8f wmma16(v16h a, v16h b, v8f c) {
  return __builtin_amdgcn_wmma_f32_16x16x32_f16(false, a, false, b, (short)0, c,
                                                false, false);
}

// ---------------------------------------------------------------------------
// Tensor Data Mover: async 2D f16 tile load (rows x 64 elements, row stride in
// elements) from global into LDS.  D# built per CDNA5 ISA sec. 8.3/8.4:
//   group0: count=1 | lds_addr | global_addr[56:0] | type=2
//   group1: data_size=2B, tensor_dim0=768, tile 64 x rows, dim0_stride
// Issued by one wave; completion via s_wait_tensorcnt + workgroup barrier.
// ---------------------------------------------------------------------------
__device__ __forceinline__ unsigned lds_off(const void* p) {
  // generic pointers to LDS carry the byte offset in the low 32 bits
  return (unsigned)(unsigned long long)(__UINTPTR_TYPE__)p;
}

__device__ __forceinline__ void tdm_load_tile_f16(unsigned ldsaddr, const void* gptr,
                                                  unsigned rows, unsigned row_stride) {
  const unsigned long long ga = (unsigned long long)(__UINTPTR_TYPE__)gptr;
  union { unsigned u[4]; u32x4 v; } g0;
  g0.u[0] = 1u;                                         // count=1, user D#
  g0.u[1] = ldsaddr;                                    // lds_addr (bytes)
  g0.u[2] = (unsigned)(ga & 0xffffffffu);               // global_addr[31:0]
  g0.u[3] = (unsigned)((ga >> 32) & 0x01ffffffu) | 0x80000000u;  // addr[56:32]|type=2
  union { unsigned u[8]; u32x8 v; } g1;
  g1.u[0] = 0x00010000u;            // wg_mask=0, data_size=1 (2 bytes), no flags
  g1.u[1] = row_stride << 16;       // tensor_dim0[15:0] (row length, elements)
  g1.u[2] = rows << 16;             // tensor_dim1[15:0]
  g1.u[3] = 64u << 16;              // tile_dim0 = 64
  g1.u[4] = rows & 0xffffu;         // tile_dim1
  g1.u[5] = row_stride;             // tensor_dim0_stride[31:0] (elements)
  g1.u[6] = 0u;
  g1.u[7] = 0u;
  asm volatile("tensor_load_to_lds %0, %1" :: "s"(g0.v), "s"(g1.v) : "memory");
}

// ---------------------------------------------------------------------------
// Kernel 1: QKV projection.  C[16384x768] = f16(X) * f16(W{q,k,v}), f16 out.
// Block tile 64(M) x 256(N), K-step 32; 8 waves, each 32x64 (8 WMMA/step).
// Q gets HD^-0.5 = 0.125 folded in.  grid = (3, 256, 3), block = 256.
// ---------------------------------------------------------------------------
__global__ __launch_bounds__(256) void qkv_gemm(
    const float* __restrict__ x, const float* __restrict__ wq,
    const float* __restrict__ wk, const float* __restrict__ wv,
    _Float16* __restrict__ ws) {
  __shared__ _Float16 As[64 * 32];    // [m][k]
  __shared__ _Float16 Bt[256 * 32];   // [n][k] (transposed)

  const int tid  = threadIdx.x;
  const int lane = tid & 31;
  const int w    = tid >> 5;
  const int m0   = blockIdx.y * 64;
  const int n0   = blockIdx.x * 256;
  const int z    = blockIdx.z;

  const float* wgt  = (z == 0) ? wq : (z == 1) ? wk : wv;
  _Float16*    out  = ws + (size_t)z * (16384ull * 768ull);
  const float scale = (z == 0) ? 0.125f : 1.0f;

  const int mrow = (w >> 2) * 32;    // 0 or 32
  const int ncol = (w & 3) * 64;     // 0,64,128,192

  v8f acc[2][4] = {};

  for (int k0 = 0; k0 < 768; k0 += 32) {
    // ---- stage A (64x32): float4 loads, packed b64 LDS stores ----
    for (int e = tid; e < 512; e += 256) {
      const int r = e >> 3, c = (e & 7) * 4;
      const float4 f = *(const float4*)(x + (size_t)(m0 + r) * 768 + k0 + c);
      union { _Float16 h[4]; unsigned long long u; } p;
      p.h[0] = (_Float16)f.x; p.h[1] = (_Float16)f.y;
      p.h[2] = (_Float16)f.z; p.h[3] = (_Float16)f.w;
      *(unsigned long long*)(As + r * 32 + c) = p.u;
    }
    // ---- stage B transposed (256n x 32k): float4 loads along n ----
    for (int e = tid; e < 2048; e += 256) {
      const int k = e >> 6, n = (e & 63) * 4;
      const float4 f = *(const float4*)(wgt + (size_t)(k0 + k) * 768 + n0 + n);
      Bt[(n + 0) * 32 + k] = (_Float16)f.x;
      Bt[(n + 1) * 32 + k] = (_Float16)f.y;
      Bt[(n + 2) * 32 + k] = (_Float16)f.z;
      Bt[(n + 3) * 32 + k] = (_Float16)f.w;
    }
    __syncthreads();

    const v16h a0 = load_frag(As, mrow, 32, 0, lane);
    const v16h a1 = load_frag(As, mrow + 16, 32, 0, lane);
#pragma unroll
    for (int j = 0; j < 4; ++j) {
      const v16h b = load_frag(Bt, ncol + 16 * j, 32, 0, lane);
      acc[0][j] = wmma16(a0, b, acc[0][j]);
      acc[1][j] = wmma16(a1, b, acc[1][j]);
    }
    __syncthreads();
  }

  const int mb = m0 + mrow + ((lane >> 4) << 3);
  const int nb = n0 + ncol + (lane & 15);
#pragma unroll
  for (int i = 0; i < 2; ++i)
#pragma unroll
    for (int j = 0; j < 4; ++j)
#pragma unroll
      for (int v = 0; v < 8; ++v)
        out[(size_t)(mb + 16 * i + v) * 768 + nb + 16 * j] =
            (_Float16)(acc[i][j][v] * scale);
}

// ---------------------------------------------------------------------------
// Kernel 2: flash attention.  One block per (b, h, 64-query chunk).
// Q/K tiles DMA'd by the Tensor Data Mover (f16, no conversion needed),
// V staged transposed with vectorized loads.  Online softmax in LDS.
// grid = (16, 12, 16), block = 256.
// ---------------------------------------------------------------------------
__global__ __launch_bounds__(256) void flash_attn(
    const _Float16* __restrict__ qg, const _Float16* __restrict__ kg,
    const _Float16* __restrict__ vg, _Float16* __restrict__ ctx) {
  __shared__ _Float16 Qs[64 * 64];   // [q][hd]
  __shared__ _Float16 Ks[64 * 64];   // [key][hd] == B^T layout for Q*K^T
  __shared__ _Float16 Vt[64 * 64];   // [hd][key] == B^T layout for P*V
  __shared__ _Float16 Ps[64 * 64];   // [q][key]
  __shared__ float    Ss[64 * 64];   // raw scores
  __shared__ float    rowm[64], rowl[64], rowc[64];

  const int tid  = threadIdx.x;
  const int lane = tid & 31;
  const int w    = tid >> 5;
  const int qt   = blockIdx.x;
  const int h    = blockIdx.y;
  const int b    = blockIdx.z;

  const size_t rowbase = ((size_t)b * 1024 + (size_t)qt * 64) * 768;
  const size_t bbase   = (size_t)b * 1024 * 768;
  const int    hoff    = h * 64;

  // Async DMA of the Q tile (issued by wave 0 only).
  if (w == 0) {
    tdm_load_tile_f16(lds_off(Qs), qg + rowbase + hoff, 64u, 768u);
    __builtin_amdgcn_s_wait_tensorcnt(0);
  }
  if (tid < 64) { rowm[tid] = -1e30f; rowl[tid] = 0.0f; rowc[tid] = 1.0f; }

  const int moff = (w >> 1) * 16;
  const int noff = (w & 1) * 32;
  v8f o0 = {}; v8f o1 = {};

  __syncthreads();

  for (int kt = 0; kt < 16; ++kt) {
    const size_t kb = bbase + (size_t)kt * 64 * 768;

    // K tile via TDM (row-major [key][hd], directly the B^T layout).
    if (w == 0) tdm_load_tile_f16(lds_off(Ks), kg + kb + hoff, 64u, 768u);

    // V tile transposed into [hd][key]: b128 loads, scattered b16 stores.
    {
      const _Float16* vsrc = vg + kb + hoff;
      for (int e = tid; e < 512; e += 256) {
        const int r = e >> 3, c = (e & 7) * 8;
        union { uint4 u; _Float16 hh[8]; } p;
        p.u = *(const uint4*)(vsrc + (size_t)r * 768 + c);
#pragma unroll
        for (int j = 0; j < 8; ++j) Vt[(c + j) * 64 + r] = p.hh[j];
      }
      if (kt + 1 < 16)  // pull next V tile toward L2 (global_prefetch)
        __builtin_prefetch(vsrc + 64 * 768 + (size_t)tid * 256, 0, 1);
    }
    if (w == 0) __builtin_amdgcn_s_wait_tensorcnt(0);
    __syncthreads();

    // ---- scores S = Q * K^T (Q already carries 1/sqrt(HD)) ----
    v8f s0 = {}; v8f s1 = {};
#pragma unroll
    for (int k0 = 0; k0 < 64; k0 += 32) {
      const v16h a  = load_frag(Qs, moff, 64, k0, lane);
      const v16h b0 = load_frag(Ks, noff, 64, k0, lane);
      const v16h b1 = load_frag(Ks, noff + 16, 64, k0, lane);
      s0 = wmma16(a, b0, s0);
      s1 = wmma16(a, b1, s1);
    }
    {  // spill raw scores
      const int mb = moff + ((lane >> 4) << 3);
      const int nb = noff + (lane & 15);
#pragma unroll
      for (int i = 0; i < 8; ++i) {
        Ss[(mb + i) * 64 + nb]      = s0[i];
        Ss[(mb + i) * 64 + nb + 16] = s1[i];
      }
    }
    __syncthreads();

    // ---- online softmax: one row per thread (tid < 64) ----
    if (tid < 64) {
      const float* sr = Ss + tid * 64;
      float mx = rowm[tid];
#pragma unroll 8
      for (int j = 0; j < 64; ++j) mx = fmaxf(mx, sr[j]);
      const float corr = __expf(rowm[tid] - mx);
      float sum = 0.0f;
      _Float16* pr = Ps + tid * 64;
#pragma unroll 8
      for (int j = 0; j < 64; ++j) {
        const float e = __expf(sr[j] - mx);
        sum += e;
        pr[j] = (_Float16)e;
      }
      rowl[tid] = rowl[tid] * corr + sum;
      rowm[tid] = mx;
      rowc[tid] = corr;
    }
    __syncthreads();

    // ---- rescale running O, then O += P * V ----
    {
      const int mb = moff + ((lane >> 4) << 3);
#pragma unroll
      for (int i = 0; i < 8; ++i) {
        const float c = rowc[mb + i];
        o0[i] *= c;
        o1[i] *= c;
      }
    }
#pragma unroll
    for (int k0 = 0; k0 < 64; k0 += 32) {
      const v16h a  = load_frag(Ps, moff, 64, k0, lane);
      const v16h b0 = load_frag(Vt, noff, 64, k0, lane);
      const v16h b1 = load_frag(Vt, noff + 16, 64, k0, lane);
      o0 = wmma16(a, b0, o0);
      o1 = wmma16(a, b1, o1);
    }
    __syncthreads();   // protect Ks/Vt/Ss/Ps/rowc before next tile
  }

  // ---- normalize and store context (f16, row-major 16384x768) ----
  {
    const int mb = moff + ((lane >> 4) << 3);
    const int nb = noff + (lane & 15);
#pragma unroll
    for (int i = 0; i < 8; ++i) {
      const float inv = 1.0f / rowl[mb + i];
      const size_t row = rowbase + (size_t)(mb + i) * 768;
      ctx[row + hoff + nb]      = (_Float16)(o0[i] * inv);
      ctx[row + hoff + nb + 16] = (_Float16)(o1[i] * inv);
    }
  }
}

// ---------------------------------------------------------------------------
// Kernel 3: output projection.  out = ctx(f16) * f16(w_o) + b_o, f32 output.
// Same tiling as kernel 1.  grid = (3, 256), block = 256.
// ---------------------------------------------------------------------------
__global__ __launch_bounds__(256) void out_gemm(
    const _Float16* __restrict__ ctx, const float* __restrict__ wo,
    const float* __restrict__ bo, float* __restrict__ out) {
  __shared__ _Float16 As[64 * 32];
  __shared__ _Float16 Bt[256 * 32];

  const int tid  = threadIdx.x;
  const int lane = tid & 31;
  const int w    = tid >> 5;
  const int m0   = blockIdx.y * 64;
  const int n0   = blockIdx.x * 256;

  const int mrow = (w >> 2) * 32;
  const int ncol = (w & 3) * 64;

  v8f acc[2][4] = {};

  for (int k0 = 0; k0 < 768; k0 += 32) {
    // A is f16 already: b128 copies (64 rows x 32 k = 256 uint4)
    for (int e = tid; e < 256; e += 256) {
      const int r = e >> 2, c = (e & 3) * 8;
      *(uint4*)(As + r * 32 + c) =
          *(const uint4*)(ctx + (size_t)(m0 + r) * 768 + k0 + c);
    }
    for (int e = tid; e < 2048; e += 256) {
      const int k = e >> 6, n = (e & 63) * 4;
      const float4 f = *(const float4*)(wo + (size_t)(k0 + k) * 768 + n0 + n);
      Bt[(n + 0) * 32 + k] = (_Float16)f.x;
      Bt[(n + 1) * 32 + k] = (_Float16)f.y;
      Bt[(n + 2) * 32 + k] = (_Float16)f.z;
      Bt[(n + 3) * 32 + k] = (_Float16)f.w;
    }
    __syncthreads();

    const v16h a0 = load_frag(As, mrow, 32, 0, lane);
    const v16h a1 = load_frag(As, mrow + 16, 32, 0, lane);
#pragma unroll
    for (int j = 0; j < 4; ++j) {
      const v16h b = load_frag(Bt, ncol + 16 * j, 32, 0, lane);
      acc[0][j] = wmma16(a0, b, acc[0][j]);
      acc[1][j] = wmma16(a1, b, acc[1][j]);
    }
    __syncthreads();
  }

  const int mb = m0 + mrow + ((lane >> 4) << 3);
  const int nb = n0 + ncol + (lane & 15);
#pragma unroll
  for (int j = 0; j < 4; ++j) {
    const float bias = bo[nb + 16 * j];
#pragma unroll
    for (int i = 0; i < 2; ++i)
#pragma unroll
      for (int v = 0; v < 8; ++v)
        out[(size_t)(mb + 16 * i + v) * 768 + nb + 16 * j] = acc[i][j][v] + bias;
  }
}

// ---------------------------------------------------------------------------
extern "C" void kernel_launch(void* const* d_in, const int* in_sizes, int n_in,
                              void* d_out, int out_size, void* d_ws, size_t ws_size,
                              hipStream_t stream) {
  const float* x  = (const float*)d_in[0];
  const float* wq = (const float*)d_in[1];
  const float* wk = (const float*)d_in[2];
  const float* wv = (const float*)d_in[3];
  const float* wo = (const float*)d_in[4];
  const float* bo = (const float*)d_in[5];
  float* out = (float*)d_out;

  _Float16* ws = (_Float16*)d_ws;
  const size_t SZ = 16384ull * 768ull;   // elements per (b*s, d) plane
  _Float16* q   = ws;
  _Float16* k   = ws + SZ;
  _Float16* v   = ws + 2 * SZ;
  _Float16* ctx = ws + 3 * SZ;           // total f16 ws: 4*SZ*2B ~= 100.7 MB

  dim3 blk(256);
  qkv_gemm <<<dim3(3, 256, 3), blk, 0, stream>>>(x, wq, wk, wv, ws);
  flash_attn<<<dim3(16, 12, 16), blk, 0, stream>>>(q, k, v, ctx);
  out_gemm <<<dim3(3, 256),    blk, 0, stream>>>(ctx, wo, bo, out);
}
